// MiniSpinNet_35407710389122
// MI455X (gfx1250) — matrix operand order repaired and checked by reading, not compiled, not run on hardware
//
#include <hip/hip_runtime.h>
#include <hip/hip_bf16.h>
#include <math.h>
#include <stdint.h>

// ---------------------------------------------------------------------------
// MiniSpinNet for MI455X (gfx1250): bf16 WMMA implicit-GEMM conv3d pipeline
// with async global->LDS activation staging, a zero-padding row in LDS for
// branch-free boundary handling, and 4x cout-blocked WMMA tiles.
// ---------------------------------------------------------------------------

typedef __attribute__((ext_vector_type(16))) __bf16 v16bf;
typedef __attribute__((ext_vector_type(8)))  float  v8f;

#define BATCH 2
#define NPTS  2048
#define MCTR  64
#define BM    128          // BATCH * MCTR
#define KBIN  512          // 8*8*8 spherical bins
#define SPAT  512          // 8x8x8 conv grid
#define ROWS  (BM * SPAT)  // 65536 GEMM rows (samples per channel for BN)
#define PI_F  3.14159265358979323846f
#define EPS_BN 1e-5f

// ---------------------------------------------------------------------------
// 1) Spherical binning + scatter-mean.  One block per (b,m) center.
//    Separable nearest-bin: idx_axis = clamp(floor(x*nbins),0,7).
// ---------------------------------------------------------------------------
__global__ __launch_bounds__(256) void bin_scatter_kernel(
    const float* __restrict__ pts, const float* __restrict__ ctr,
    float* __restrict__ binf)
{
    __shared__ float sS[KBIN * 3];
    __shared__ float sC[KBIN];
    const int bm = blockIdx.x;
    const int b = bm >> 6, m = bm & 63;
    const int tid = threadIdx.x;
    for (int i = tid; i < KBIN * 3; i += 256) sS[i] = 0.0f;
    for (int i = tid; i < KBIN; i += 256) sC[i] = 0.0f;
    __syncthreads();
    const float cx = ctr[(b * MCTR + m) * 3 + 0];
    const float cy = ctr[(b * MCTR + m) * 3 + 1];
    const float cz = ctr[(b * MCTR + m) * 3 + 2];
    for (int i = tid; i < NPTS; i += 256) {
        const float* p = pts + (b * NPTS + i) * 3;
        float rx = p[0] - cx, ry = p[1] - cy, rz = p[2] - cz;
        float rho = sqrtf(rx * rx + ry * ry + rz * rz);
        float th = acosf(rz / (rho + 1e-8f));
        float ph = atan2f(ry, rx);
        int ir = min(max((int)floorf(rho * 16.0f), 0), 7);          // (rho/0.5)*8
        int it = min(max((int)floorf(th * (8.0f / PI_F)), 0), 7);   // theta/(pi/8)
        int ip = min(max((int)floorf((ph + PI_F) * (4.0f / PI_F)), 0), 7);
        int k = (ir << 6) | (it << 3) | ip;
        atomicAdd(&sS[k * 3 + 0], rx);
        atomicAdd(&sS[k * 3 + 1], ry);
        atomicAdd(&sS[k * 3 + 2], rz);
        atomicAdd(&sC[k], 1.0f);
    }
    __syncthreads();
    for (int k = tid; k < KBIN; k += 256) {
        float inv = 1.0f / fmaxf(sC[k], 1.0f);
        binf[(bm * KBIN + k) * 3 + 0] = sS[k * 3 + 0] * inv;
        binf[(bm * KBIN + k) * 3 + 1] = sS[k * 3 + 1] * inv;
        binf[(bm * KBIN + k) * 3 + 2] = sS[k * 3 + 2] * inv;
    }
}

// ---------------------------------------------------------------------------
// 2) xyz-raising 1x1 conv (3 -> 32), channel-minor output [row][c], f32.
// ---------------------------------------------------------------------------
__global__ __launch_bounds__(256) void raise_kernel(
    const float* __restrict__ binf, const float* __restrict__ wr,
    const float* __restrict__ br, float* __restrict__ x0)
{
    int i = blockIdx.x * 256 + threadIdx.x;       // ROWS*32 threads
    int c = i & 31, row = i >> 5;
    float v0 = binf[row * 3 + 0], v1 = binf[row * 3 + 1], v2 = binf[row * 3 + 2];
    x0[i] = wr[c * 3 + 0] * v0 + wr[c * 3 + 1] * v1 + wr[c * 3 + 2] * v2 + br[c];
}

// ---------------------------------------------------------------------------
// BN stats helpers: per-channel sum / sumsq over ROWS samples, channel-minor.
// ---------------------------------------------------------------------------
__global__ void zero_stats_kernel(float* __restrict__ stats)
{
    stats[threadIdx.x] = 0.0f;                    // 512 threads
}

__global__ __launch_bounds__(256) void reduce_stats_kernel(
    const float* __restrict__ y, float* __restrict__ stats, int C, int total)
{
    __shared__ float s[512];
    int tid = threadIdx.x;
    s[tid] = 0.0f; s[tid + 256] = 0.0f;
    __syncthreads();
    float sum = 0.0f, sq = 0.0f;
    for (int i = blockIdx.x * 256 + tid; i < total; i += 256 * gridDim.x) {
        float v = y[i];
        sum += v; sq += v * v;
    }
    int c = tid & (C - 1);
    atomicAdd(&s[c], sum);
    atomicAdd(&s[C + c], sq);
    __syncthreads();
    if (tid < C) {
        atomicAdd(&stats[tid], s[tid]);
        atomicAdd(&stats[C + tid], s[C + tid]);
    }
}

// ---------------------------------------------------------------------------
// BN (training-mode, biased var) + ReLU.  Writes bf16 activations for the next
// WMMA conv, or f32 in-place for the final layer.
// ---------------------------------------------------------------------------
__global__ __launch_bounds__(256) void bnrelu_kernel(
    const float* __restrict__ y, const float* __restrict__ stats,
    const float* __restrict__ g, const float* __restrict__ be,
    __bf16* __restrict__ outb, float* __restrict__ outf,
    int C, float invRows, int total)
{
    int i = blockIdx.x * 256 + threadIdx.x;
    if (i >= total) return;
    int c = i & (C - 1);
    float mean = stats[c] * invRows;
    float var = stats[C + c] * invRows - mean * mean;
    float scale = g[c] * rsqrtf(var + EPS_BN);
    float val = fmaxf(0.0f, (y[i] - mean) * scale + be[c]);
    if (outb) outb[i] = (__bf16)val;
    else      outf[i] = val;
}

// ---------------------------------------------------------------------------
// Weight pre-pack: f32 OIDHW -> bf16 in exact WMMA A-fragment order so the
// conv kernel's A operand is a single 32-byte (v16bf) load per K-chunk.
// 16x32 bf16 A layout: lane l (l16=l&15,hi=l>>4) holds row M=l16;
// element m maps to k = hi*16 + m (contiguous, matches B).
// K ordering: outer tap t, inner ci chunks of 32 (B side uses identity k->ci).
// ---------------------------------------------------------------------------
__global__ __launch_bounds__(256) void wprep_kernel(
    const float* __restrict__ wsrc, __bf16* __restrict__ wP,
    int Cin, int Cout, int total)
{
    int e = blockIdx.x * 256 + threadIdx.x;
    if (e >= total) return;
    int m = e & 15, lane = (e >> 4) & 31;
    int rest = e >> 9;
    int nCoT = Cout >> 4, nCiC = Cin >> 5;
    int coT = rest % nCoT; rest /= nCoT;
    int ciC = rest % nCiC;
    int t = rest / nCiC;
    int hi = lane >> 4, l16 = lane & 15;
    int kA = hi * 16 + m;
    int co = (coT << 4) + l16;
    int ci = (ciC << 5) + kA;
    wP[e] = (__bf16)wsrc[(co * Cin + ci) * 27 + t];
}

// ---------------------------------------------------------------------------
// 3x3x3 conv3d (pad 1) as implicit GEMM on v_wmma_f32_16x16x32_bf16.
// One 256-thread block (8 waves) per bm: the whole 8x8x8 x Cin activation
// slab (+ one zeroed pad row at slab[SPAT]) is staged into LDS via gfx1250
// async global->LDS loads.  Out-of-bounds taps are redirected to the pad row
// by an integer select, so the K-loop is branch-free: each K-chunk does
// one 32-byte LDS load (B, reused 4x) + 4 pre-packed A loads + 4 WMMAs.
// ---------------------------------------------------------------------------
__global__ __launch_bounds__(256) void conv3d_wmma_kernel(
    const __bf16* __restrict__ act, const __bf16* __restrict__ wP,
    const float* __restrict__ bias, float* __restrict__ out,
    int Cin, int Cout)
{
    extern __shared__ __bf16 slab[];              // (SPAT+1) * Cin bf16
    const int bm = blockIdx.x;
    const int coG = blockIdx.y;                   // group of 64 output channels
    const int tid = threadIdx.x;
    const int lane = tid & 31, wave = tid >> 5;
    const int l16 = lane & 15, hi = lane >> 4;
    const int nCiC = Cin >> 5;
    const int nCoT = Cout >> 4;

    // ---- stage activation slab: global -> LDS via async copy (ASYNCcnt) ----
    {
        unsigned ldsBase = (unsigned)(uintptr_t)&slab[0];
        const __bf16* gsrc = act + bm * SPAT * Cin;
        const int nv = (SPAT * Cin) >> 3;         // number of 16B vectors
        for (int v = tid; v < nv; v += 256) {
            unsigned laddr = ldsBase + (v << 4);
            const __bf16* g = gsrc + (v << 3);
            asm volatile("global_load_async_to_lds_b128 %0, %1, off"
                         :: "v"(laddr), "v"(g) : "memory");
        }
        // zero-pad row for out-of-bounds taps (ds stores)
        for (int i = tid; i < Cin; i += 256) slab[SPAT * Cin + i] = (__bf16)0.0f;
        asm volatile("s_wait_asynccnt 0" ::: "memory");
    }
    __syncthreads();

    // 32 spatial tiles per bm; 8 waves -> 4 tiles each, 4 coTiles per tile.
    for (int st = wave; st < 32; st += 8) {
        const int pos = (st << 4) + l16;          // this lane's output column
        const int d = pos >> 6, h = (pos >> 3) & 7, w = pos & 7;
        v8f acc[4] = {{}, {}, {}, {}};

        for (int t = 0; t < 27; ++t) {
            int kd = t / 9, rem = t % 9, kh = rem / 3, kw = rem % 3;
            int sd = d + kd - 1, sh = h + kh - 1, sw = w + kw - 1;
            bool valid = ((unsigned)sd < 8u) & ((unsigned)sh < 8u) & ((unsigned)sw < 8u);
            // invalid taps read the zeroed pad row -> branch-free inner loop
            int srow = valid ? ((sd << 6) | (sh << 3) | sw) : SPAT;
            const __bf16* brow = slab + srow * Cin + (hi << 4);
            for (int cc = 0; cc < nCiC; ++cc) {
                v16bf bfrag = *(const v16bf*)(brow + (cc << 5));
                const __bf16* wp = wP +
                    (((((t * nCiC + cc) * nCoT + (coG << 2)) << 5) | lane) << 4);
                v16bf a0 = *(const v16bf*)(wp + 0);
                v16bf a1 = *(const v16bf*)(wp + 512);
                v16bf a2 = *(const v16bf*)(wp + 1024);
                v16bf a3 = *(const v16bf*)(wp + 1536);
                acc[0] = __builtin_amdgcn_wmma_f32_16x16x32_bf16(
                    false, a0, false, bfrag, (short)0, acc[0], false, false);
                acc[1] = __builtin_amdgcn_wmma_f32_16x16x32_bf16(
                    false, a1, false, bfrag, (short)0, acc[1], false, false);
                acc[2] = __builtin_amdgcn_wmma_f32_16x16x32_bf16(
                    false, a2, false, bfrag, (short)0, acc[2], false, false);
                acc[3] = __builtin_amdgcn_wmma_f32_16x16x32_bf16(
                    false, a3, false, bfrag, (short)0, acc[3], false, false);
            }
        }
        // C/D layout: lane (hi,l16), VGPR v -> row co = coT*16 + 8*hi + v.
#pragma unroll
        for (int q = 0; q < 4; ++q) {
            int co0 = (((coG << 2) + q) << 4) + (hi << 3);
            float* op = out + (bm * SPAT + pos) * Cout + co0;
            float4 r0 = { acc[q][0] + bias[co0 + 0], acc[q][1] + bias[co0 + 1],
                          acc[q][2] + bias[co0 + 2], acc[q][3] + bias[co0 + 3] };
            float4 r1 = { acc[q][4] + bias[co0 + 4], acc[q][5] + bias[co0 + 5],
                          acc[q][6] + bias[co0 + 6], acc[q][7] + bias[co0 + 7] };
            *(float4*)(op + 0) = r0;
            *(float4*)(op + 4) = r1;
        }
    }
}

// ---------------------------------------------------------------------------
// Final spatial max over 512 positions -> (BM, 256) f32.
// ---------------------------------------------------------------------------
__global__ __launch_bounds__(256) void maxpool_kernel(
    const float* __restrict__ y, float* __restrict__ out)
{
    int bm = blockIdx.x, c = threadIdx.x;         // 256 channels
    const float* p = y + bm * SPAT * 256 + c;
    float mx = -3.4e38f;
    for (int s = 0; s < SPAT; ++s) mx = fmaxf(mx, p[s * 256]);
    out[bm * 256 + c] = mx;
}

// ---------------------------------------------------------------------------
extern "C" void kernel_launch(void* const* d_in, const int* in_sizes, int n_in,
                              void* d_out, int out_size, void* d_ws, size_t ws_size,
                              hipStream_t stream)
{
    const float* points  = (const float*)d_in[0];
    const float* centers = (const float*)d_in[1];
    const float* w_raise = (const float*)d_in[2];
    const float* b_raise = (const float*)d_in[3];
    const float* g0  = (const float*)d_in[4];
    const float* be0 = (const float*)d_in[5];
    const float* w1  = (const float*)d_in[6];
    const float* b1  = (const float*)d_in[7];
    const float* g1  = (const float*)d_in[8];
    const float* be1 = (const float*)d_in[9];
    const float* w2  = (const float*)d_in[10];
    const float* b2  = (const float*)d_in[11];
    const float* g2  = (const float*)d_in[12];
    const float* be2 = (const float*)d_in[13];
    const float* w3  = (const float*)d_in[14];
    const float* b3  = (const float*)d_in[15];
    const float* g3  = (const float*)d_in[16];
    const float* be3 = (const float*)d_in[17];

    // ---- workspace carve-up (256B aligned chunks) ----
    size_t off = 0;
    auto carve = [&](size_t bytes) -> char* {
        char* p = (char*)d_ws + off;
        off = (off + bytes + 255) & ~(size_t)255;
        return p;
    };
    float*  binf  = (float*)carve((size_t)BM * KBIN * 3 * 4);
    float*  x0    = (float*)carve((size_t)ROWS * 32 * 4);
    __bf16* act0  = (__bf16*)carve((size_t)ROWS * 32 * 2);
    float*  y1    = (float*)carve((size_t)ROWS * 64 * 4);
    __bf16* act1  = (__bf16*)carve((size_t)ROWS * 64 * 2);
    float*  y2    = (float*)carve((size_t)ROWS * 128 * 4);
    __bf16* act2  = (__bf16*)carve((size_t)ROWS * 128 * 2);
    float*  y3    = (float*)carve((size_t)ROWS * 256 * 4);
    __bf16* wP1   = (__bf16*)carve((size_t)27 * 32 * 64 * 2);
    __bf16* wP2   = (__bf16*)carve((size_t)27 * 64 * 128 * 2);
    __bf16* wP3   = (__bf16*)carve((size_t)27 * 128 * 256 * 2);
    float*  stats = (float*)carve(512 * 4);

    const float invRows = 1.0f / (float)ROWS;

    // 1) binning + scatter-mean
    bin_scatter_kernel<<<BM, 256, 0, stream>>>(points, centers, binf);

    // 2) xyz raise (3->32) + BN0 + ReLU -> bf16 act0
    raise_kernel<<<(ROWS * 32) / 256, 256, 0, stream>>>(binf, w_raise, b_raise, x0);
    zero_stats_kernel<<<1, 512, 0, stream>>>(stats);
    reduce_stats_kernel<<<1024, 256, 0, stream>>>(x0, stats, 32, ROWS * 32);
    bnrelu_kernel<<<(ROWS * 32) / 256, 256, 0, stream>>>(
        x0, stats, g0, be0, act0, nullptr, 32, invRows, ROWS * 32);

    // 3) conv1: 32 -> 64   (LDS slab: 513*32*2 ~= 32 KB)
    wprep_kernel<<<(27 * 32 * 64 + 255) / 256, 256, 0, stream>>>(w1, wP1, 32, 64, 27 * 32 * 64);
    conv3d_wmma_kernel<<<dim3(BM, 1), 256, (SPAT + 1) * 32 * 2, stream>>>(act0, wP1, b1, y1, 32, 64);
    zero_stats_kernel<<<1, 512, 0, stream>>>(stats);
    reduce_stats_kernel<<<1024, 256, 0, stream>>>(y1, stats, 64, ROWS * 64);
    bnrelu_kernel<<<(ROWS * 64) / 256, 256, 0, stream>>>(
        y1, stats, g1, be1, act1, nullptr, 64, invRows, ROWS * 64);

    // 4) conv2: 64 -> 128  (LDS slab ~= 64 KB)
    wprep_kernel<<<(27 * 64 * 128 + 255) / 256, 256, 0, stream>>>(w2, wP2, 64, 128, 27 * 64 * 128);
    conv3d_wmma_kernel<<<dim3(BM, 2), 256, (SPAT + 1) * 64 * 2, stream>>>(act1, wP2, b2, y2, 64, 128);
    zero_stats_kernel<<<1, 512, 0, stream>>>(stats);
    reduce_stats_kernel<<<1024, 256, 0, stream>>>(y2, stats, 128, ROWS * 128);
    bnrelu_kernel<<<(ROWS * 128) / 256, 256, 0, stream>>>(
        y2, stats, g2, be2, act2, nullptr, 128, invRows, ROWS * 128);

    // 5) conv3: 128 -> 256 (LDS slab ~= 128 KB), BN+ReLU in-place f32
    wprep_kernel<<<(27 * 128 * 256 + 255) / 256, 256, 0, stream>>>(w3, wP3, 128, 256, 27 * 128 * 256);
    conv3d_wmma_kernel<<<dim3(BM, 4), 256, (SPAT + 1) * 128 * 2, stream>>>(act2, wP3, b3, y3, 128, 256);
    zero_stats_kernel<<<1, 512, 0, stream>>>(stats);
    reduce_stats_kernel<<<1024, 256, 0, stream>>>(y3, stats, 256, ROWS * 256);
    bnrelu_kernel<<<(ROWS * 256) / 256, 256, 0, stream>>>(
        y3, stats, g3, be3, nullptr, y3, 256, invRows, ROWS * 256);

    // 6) spatial max -> (B, M, 256)
    maxpool_kernel<<<BM, 256, 0, stream>>>(y3, (float*)d_out);
}
